// CorticalColumnGroup_67199058313498
// MI455X (gfx1250) — compile-verified, matrix operand order checked by reading, not run on hardware
//
#include <hip/hip_runtime.h>
#include <math.h>

// ---------------------------------------------------------------------------
// CorticalColumnGroup fused bf16-WMMA implementation for gfx1250 (MI455X).
// C=16 columns, DCOL=256, DMEM=128, DPCM=64, EXP=4, tokens T = B*N = 4096.
// ---------------------------------------------------------------------------

#define CCOL 16
#define TOK  4096

typedef __attribute__((ext_vector_type(16))) __bf16 v16bf;
typedef __attribute__((ext_vector_type(8)))  float  v8f;
typedef __attribute__((ext_vector_type(8)))  unsigned int v8u_t;
typedef __attribute__((ext_vector_type(4)))  unsigned int u32x4;
typedef __attribute__((ext_vector_type(4)))  int i32x4;
typedef __attribute__((ext_vector_type(8)))  int i32x8;

#if __has_builtin(__builtin_amdgcn_tensor_load_to_lds)
#define HAVE_TDM 1
#else
#define HAVE_TDM 0
#endif

__device__ __forceinline__ unsigned short f2bf(float f) {
  unsigned u = __builtin_bit_cast(unsigned, f);
  unsigned r = u + 0x7FFFu + ((u >> 16) & 1u);   // round-to-nearest-even
  return (unsigned short)(r >> 16);
}
__device__ __forceinline__ float bf2f(unsigned short h) {
  unsigned u = ((unsigned)h) << 16;
  return __builtin_bit_cast(float, u);
}

// Build one 16x32 bf16 WMMA fragment (A or B) from k-contiguous bf16 pairs.
// Row base pointer p is in uint (bf16-pair) units; ISA layout (CDNA5 ISA):
//   VGPR v holds K pair at ku = (v&3) + half*4 + (v>>2)*8  (half = lane>>4).
__device__ __forceinline__ v16bf load_frag(const unsigned int* p, int ks, int half) {
  v8u_t u;
#pragma unroll
  for (int v = 0; v < 8; ++v) {
    int ku = (v & 3) + half * 4 + ((v >> 2) << 3);
    u[v] = p[ks * 16 + ku];
  }
  return __builtin_bit_cast(v16bf, u);
}

#if HAVE_TDM
// TDM: DMA `rows` x K bf16 (k-contiguous rows, row stride K elements) from
// global into LDS, padding 1 DWORD after each row of K/2 DWORDs so fragment
// ds reads land on distinct banks.  LDS row stride = K/2 + 1 uints.
// Issue from one wave, then s_wait_tensorcnt 0 before the block barrier.
__device__ __forceinline__ void tdm_load_rows(const unsigned int* gsrc,
                                              unsigned int* lds, int K, int rows) {
  unsigned lds_off = (unsigned)(uintptr_t)(void*)lds;
  unsigned long long ga = (unsigned long long)(uintptr_t)(const void*)gsrc;
  unsigned pcode = (unsigned)(__builtin_ctz((unsigned)K) - 2);  // K/2 DWORDs/row
  u32x4 g0;
  g0[0] = 1u;                                    // count=1, user descriptor
  g0[1] = lds_off;                               // lds_addr
  g0[2] = (unsigned)(ga & 0xffffffffu);          // global_addr[31:0]
  g0[3] = (unsigned)((ga >> 32) & 0x1ffffffu) | (2u << 30);  // addr[56:32] | type=2
  i32x8 g1;
  g1[0] = (int)((1u << 16) | (1u << 20) | (pcode << 22));  // data_size=2B, pad_en
  g1[1] = (int)((unsigned)K << 16);              // tensor_dim0 = K
  g1[2] = 0;
  g1[3] = (int)(((unsigned)K << 16) | 1u);       // tensor_dim1=65536, tile_dim0=K
  g1[4] = rows;                                  // tile_dim1 = rows, tile_dim2 = 0
  g1[5] = K;                                     // tensor_dim0_stride = K
  g1[6] = (int)((unsigned)K << 16);              // tensor_dim1_stride = K
  g1[7] = 0;
  i32x4 gz = {0, 0, 0, 0};
#if __clang_major__ >= 23
  i32x8 gz8 = {0, 0, 0, 0, 0, 0, 0, 0};
  __builtin_amdgcn_tensor_load_to_lds(g0, g1, gz, gz, gz8, 0);
#else
  __builtin_amdgcn_tensor_load_to_lds(g0, g1, gz, gz, 0);
#endif
#if __has_builtin(__builtin_amdgcn_s_wait_tensorcnt)
  __builtin_amdgcn_s_wait_tensorcnt(0);
#endif
}
#endif

// ---------------------------------------------------------------------------
// Generic batched (per-column) GEMM: D[t][c][n] = A[c][t][:K] * Wt[c][n][:K]
//   A  : bf16 pairs, row-major [C][TOK][K/2]   (uint units)
//   Wt : bf16 pairs, transposed weights [C][Ntot][K/2]
// Block = 256 threads = 8 waves (2 M-tiles x 4 N-tiles of 16x16).
// A tile (32 x K) staged into LDS via the Tensor Data Mover (one issue per
// block) instead of 4x-redundant per-wave global reads.
// EPI 0: out_f32[(t*CCOL+c)*Ntot + n] = acc + bias
// EPI 1: h = LN(x) * (1 + 0.1*tanh(acc+bias)) -> bf16   (gain path; LN(x)
//        precomputed in bf16 by xprep_k)
// ---------------------------------------------------------------------------
template <int EPI>
__global__ __launch_bounds__(256) void gemm_k(
    const unsigned int* Au, const unsigned int* Wtu, const float* bias,
    int K, int Ntot, float* outF, unsigned short* outH,
    const unsigned short* lnxH) {
  const int lane = threadIdx.x & 31, wid = threadIdx.x >> 5;
  const int wm = wid >> 2, wn = wid & 3;
  const int half = lane >> 4, rl = lane & 15;
  const int c = blockIdx.z;
  const int t0 = blockIdx.x * 32 + wm * 16;
  const int n0 = blockIdx.y * 64 + wn * 16;
  const int K2 = K >> 1;
#if HAVE_TDM
  __shared__ unsigned int a_lds[32 * 129];       // K <= 256 -> stride K2+1 <= 129
  if (wid == 0)
    tdm_load_rows(Au + ((size_t)c * TOK + blockIdx.x * 32) * K2, a_lds, K, 32);
  __syncthreads();
  const unsigned int* Ap = a_lds + (wm * 16 + rl) * (K2 + 1);
#else
  const unsigned int* Ap = Au + ((size_t)c * TOK + (t0 + rl)) * K2;
#endif
  const unsigned int* Bp = Wtu + ((size_t)c * Ntot + (n0 + rl)) * K2;
  v8f acc = {};
  const int ksteps = K >> 5;
  for (int ks = 0; ks < ksteps; ++ks) {
    __builtin_prefetch(Bp + (ks + 1) * 16, 0, 1);
    v16bf a = load_frag(Ap, ks, half);
    v16bf b = load_frag(Bp, ks, half);
    acc = __builtin_amdgcn_wmma_f32_16x16x32_bf16(false, a, false, b,
                                                  (short)0, acc, false, false);
  }
  const int col = n0 + rl;
#pragma unroll
  for (int r = 0; r < 8; ++r) {
    const int tt = t0 + r + half * 8;
    float val = acc[r] + bias[c * Ntot + col];
    if constexpr (EPI == 0) {
      outF[((size_t)tt * CCOL + c) * Ntot + col] = val;
    } else {
      float g = 1.0f + 0.1f * tanhf(val);
      float lx = bf2f(lnxH[((size_t)c * TOK + tt) * 256 + col]);
      outH[((size_t)c * TOK + tt) * 256 + col] = f2bf(lx * g);
    }
  }
}

// ---------------------------------------------------------------------------
// Fused FFN: x_out = x + down( gelu_erf( up(h) + b_up ) ) + b_dn
// Block = 16 tokens x 1 column, 8 waves. h-tile via TDM into LDS; the
// 16x1024 GELU intermediate also lives entirely in LDS (rows of 514 uints).
// ---------------------------------------------------------------------------
__global__ __launch_bounds__(256) void ffn_k(
    const unsigned int* Hu,      // h bf16 pairs [C][TOK][128]
    const unsigned int* UpTu,    // up^T  bf16 pairs [C][1024][128]
    const unsigned int* DnTu,    // down^T bf16 pairs [C][256][512]
    const float* upb, const float* dnb, const float* xF,
    float* outX, unsigned short* outXob) {
  __shared__ unsigned int u_lds[16 * 514];
  const int lane = threadIdx.x & 31, wid = threadIdx.x >> 5;
  const int half = lane >> 4, rl = lane & 15;
  const int c = blockIdx.y;
  const int t0 = blockIdx.x * 16;

#if HAVE_TDM
  __shared__ unsigned int h_lds[16 * 129];
  if (wid == 0)
    tdm_load_rows(Hu + ((size_t)c * TOK + t0) * 128, h_lds, 256, 16);
  __syncthreads();
  const unsigned int* Ap = h_lds + rl * 129;
#else
  const unsigned int* Ap = Hu + ((size_t)c * TOK + (t0 + rl)) * 128;
#endif

  // Preload the 16x256 A (h) fragments once; reused for all 64 hidden tiles.
  v16bf afr[8];
#pragma unroll
  for (int ks = 0; ks < 8; ++ks) afr[ks] = load_frag(Ap, ks, half);

  // Phase 1: up-proj + exact-erf GELU -> LDS
  unsigned short* us = (unsigned short*)u_lds;
  for (int j = 0; j < 8; ++j) {
    const int n0 = wid * 128 + j * 16;
    const unsigned int* Bp = UpTu + ((size_t)c * 1024 + (n0 + rl)) * 128;
    v8f acc = {};
#pragma unroll
    for (int ks = 0; ks < 8; ++ks) {
      v16bf b = load_frag(Bp, ks, half);
      acc = __builtin_amdgcn_wmma_f32_16x16x32_bf16(false, afr[ks], false, b,
                                                    (short)0, acc, false, false);
    }
#pragma unroll
    for (int r = 0; r < 8; ++r) {
      const int m = r + half * 8;
      const int col = n0 + rl;
      float v = acc[r] + upb[c * 1024 + col];
      float g = 0.5f * v * (1.0f + erff(v * 0.70710678118f));
      us[m * 1028 + col] = f2bf(g);
    }
  }
  __syncthreads();

  // Phase 2: down-proj (K=1024 from LDS) + bias + residual
  for (int j = 0; j < 2; ++j) {
    const int n0 = wid * 32 + j * 16;
    const unsigned int* Bp = DnTu + ((size_t)c * 256 + (n0 + rl)) * 512;
    const unsigned int* Aup = u_lds + rl * 514;
    v8f acc = {};
    for (int ks = 0; ks < 32; ++ks) {
      v16bf a = load_frag(Aup, ks, half);
      v16bf b = load_frag(Bp, ks, half);
      acc = __builtin_amdgcn_wmma_f32_16x16x32_bf16(false, a, false, b,
                                                    (short)0, acc, false, false);
    }
#pragma unroll
    for (int r = 0; r < 8; ++r) {
      const int tt = t0 + r + half * 8;
      const int col = n0 + rl;
      const size_t gi = ((size_t)tt * CCOL + c) * 256 + col;
      float v = acc[r] + dnb[c * 256 + col] + xF[gi];
      outX[gi] = v;
      outXob[((size_t)c * TOK + tt) * 256 + col] = f2bf(v);
    }
  }
}

// ---------------------------------------------------------------------------
// Fused heads: the four 256->128 projections (kpre/vpost/kcand/vcand) share
// the same A (x_out bf16): TDM-stage the 32x256 A tile once, preload the 8 A
// fragments, loop over the 4 weight sets.
// ---------------------------------------------------------------------------
__global__ __launch_bounds__(256) void heads_k(
    const unsigned int* Au,
    const unsigned int* W0, const unsigned int* W1,
    const unsigned int* W2, const unsigned int* W3,
    const float* b0, const float* b1, const float* b2, const float* b3,
    float* o0, float* o1, float* o2, float* o3) {
  const int lane = threadIdx.x & 31, wid = threadIdx.x >> 5;
  const int wm = wid >> 2, wn = wid & 3;
  const int half = lane >> 4, rl = lane & 15;
  const int c = blockIdx.z;
  const int t0 = blockIdx.x * 32 + wm * 16;
  const int n0 = blockIdx.y * 64 + wn * 16;
#if HAVE_TDM
  __shared__ unsigned int a_lds[32 * 129];
  if (wid == 0)
    tdm_load_rows(Au + ((size_t)c * TOK + blockIdx.x * 32) * 128, a_lds, 256, 32);
  __syncthreads();
  const unsigned int* Ap = a_lds + (wm * 16 + rl) * 129;
#else
  const unsigned int* Ap = Au + ((size_t)c * TOK + (t0 + rl)) * 128;
#endif
  v16bf afr[8];
#pragma unroll
  for (int ks = 0; ks < 8; ++ks) afr[ks] = load_frag(Ap, ks, half);
  const unsigned int* Ws[4] = {W0, W1, W2, W3};
  const float* Bs[4] = {b0, b1, b2, b3};
  float* Os[4] = {o0, o1, o2, o3};
  const int col = n0 + rl;
#pragma unroll
  for (int w = 0; w < 4; ++w) {
    const unsigned int* Bp = Ws[w] + ((size_t)c * 128 + (n0 + rl)) * 128;
    v8f acc = {};
#pragma unroll
    for (int ks = 0; ks < 8; ++ks) {
      v16bf b = load_frag(Bp, ks, half);
      acc = __builtin_amdgcn_wmma_f32_16x16x32_bf16(false, afr[ks], false, b,
                                                    (short)0, acc, false, false);
    }
#pragma unroll
    for (int r = 0; r < 8; ++r) {
      const int tt = t0 + r + half * 8;
      Os[w][((size_t)tt * CCOL + c) * 128 + col] = acc[r] + Bs[w][c * 128 + col];
    }
  }
}

// ------------------------- small VALU helper kernels -----------------------

// Fused x prep: one read of x fp32 -> LN stats -> write x bf16 AND fully
// applied LN(x) bf16 (affine folded). One wave per (c,t) row.
__global__ __launch_bounds__(256) void xprep_k(const float* xF, const float* lnw,
                                               const float* lnb,
                                               unsigned short* xb,
                                               unsigned short* lnxb) {
  const int lane = threadIdx.x & 31, wid = threadIdx.x >> 5;
  const int tc = blockIdx.x * 8 + wid;
  const int c = tc >> 12, t = tc & 4095;
  const float* row = xF + ((size_t)t * CCOL + c) * 256;
  float v[8];
  float s = 0.f, q = 0.f;
#pragma unroll
  for (int i = 0; i < 8; ++i) { v[i] = row[lane + i * 32]; s += v[i]; q += v[i] * v[i]; }
  for (int off = 16; off; off >>= 1) { s += __shfl_xor(s, off, 32); q += __shfl_xor(q, off, 32); }
  float mean = s * (1.0f / 256.0f);
  float var = q * (1.0f / 256.0f) - mean * mean;
  float rstd = rsqrtf(var + 1e-5f);
  unsigned short* xrow = xb + ((size_t)c * TOK + t) * 256;
  unsigned short* lxrow = lnxb + ((size_t)c * TOK + t) * 256;
#pragma unroll
  for (int i = 0; i < 8; ++i) {
    int d = lane + i * 32;
    xrow[d] = f2bf(v[i]);
    lxrow[d] = f2bf((v[i] - mean) * rstd * lnw[c * 256 + d] + lnb[c * 256 + d]);
  }
}

// delta = z - z_hat_prev (bf16 out for gain GEMM), surprise, gate.
__global__ __launch_bounds__(256) void delta_k(const float* zF, const float* zhp,
                                               unsigned short* deltab,
                                               float* surp, float* gate) {
  const int lane = threadIdx.x & 31, wid = threadIdx.x >> 5;
  const int tc = blockIdx.x * 8 + wid;
  const int c = tc >> 12, t = tc & 4095;
  const size_t base = ((size_t)t * CCOL + c) * 64;
  float ss = 0.f;
#pragma unroll
  for (int i = 0; i < 2; ++i) {
    int d = lane + i * 32;
    float dv = zF[base + d] - zhp[base + d];
    deltab[((size_t)c * TOK + t) * 64 + d] = f2bf(dv);
    ss += dv * dv;
  }
  for (int off = 16; off; off >>= 1) ss += __shfl_xor(ss, off, 32);
  if (lane == 0) {
    float s = ss * (1.0f / 64.0f);
    surp[t * CCOL + c] = s;
    gate[t * CCOL + c] = fminf(fmaxf(s, 0.0f), 1.0f);
  }
}

// fp32 W[c][K][N] -> bf16 W^T [c][N][K] (k-contiguous pairs for WMMA frags)
__global__ void wconv_k(const float* W, unsigned short* Wt, int Kd, int Nd) {
  const size_t total = (size_t)CCOL * Kd * Nd;
  for (size_t i = (size_t)blockIdx.x * blockDim.x + threadIdx.x; i < total;
       i += (size_t)gridDim.x * blockDim.x) {
    int n = (int)(i % Nd);
    size_t r = i / Nd;
    int k = (int)(r % Kd);
    int c = (int)(r / Kd);
    Wt[((size_t)c * Nd + n) * Kd + k] = f2bf(W[i]);
  }
}

// z[t][c][64] fp32 (from d_out) -> zb[c][t][64] bf16
__global__ void zconv_k(const float* z, unsigned short* zb) {
  const size_t total = (size_t)TOK * CCOL * 64;
  for (size_t i = (size_t)blockIdx.x * blockDim.x + threadIdx.x; i < total;
       i += (size_t)gridDim.x * blockDim.x) {
    int d = (int)(i & 63); int c = (int)((i >> 6) & 15); size_t t = i >> 10;
    zb[((size_t)c * TOK + t) * 64 + d] = f2bf(z[i]);
  }
}

// In-place L2 unit-normalize rows of 128 (k_cand / q_nov); wave per token-col.
__global__ __launch_bounds__(256) void norm_k(float* reg) {
  const int lane = threadIdx.x & 31, wid = threadIdx.x >> 5;
  const int tc = blockIdx.x * 8 + wid;
  const int c = tc >> 12, t = tc & 4095;
  float* row = reg + ((size_t)t * CCOL + c) * 128;
  float v[4]; float ss = 0.f;
#pragma unroll
  for (int i = 0; i < 4; ++i) { v[i] = row[lane + i * 32]; ss += v[i] * v[i]; }
  for (int off = 16; off; off >>= 1) ss += __shfl_xor(ss, off, 32);
  float inv = 1.0f / fmaxf(sqrtf(ss), 1e-12f);
#pragma unroll
  for (int i = 0; i < 4; ++i) row[lane + i * 32] = v[i] * inv;
}

// w_nov = sigmoid(x_out . nov_w + nov_b); reads the bf16 x_out copy (32MB
// instead of 64MB fp32); wave per token-col (N=1 head).
__global__ __launch_bounds__(256) void nov_k(const unsigned short* xob,
                                             const float* nw, const float* nb,
                                             float* wout) {
  const int lane = threadIdx.x & 31, wid = threadIdx.x >> 5;
  const int tc = blockIdx.x * 8 + wid;
  const int c = tc >> 12, t = tc & 4095;
  const unsigned short* row = xob + ((size_t)c * TOK + t) * 256;
  const float* w = nw + c * 256;
  float s = 0.f;
#pragma unroll
  for (int i = 0; i < 8; ++i) s += bf2f(row[lane + i * 32]) * w[lane + i * 32];
  for (int off = 16; off; off >>= 1) s += __shfl_xor(s, off, 32);
  if (lane == 0) wout[t * CCOL + c] = 1.0f / (1.0f + expf(-(s + nb[c])));
}

// ---------------------------------------------------------------------------
extern "C" void kernel_launch(void* const* d_in, const int* in_sizes, int n_in,
                              void* d_out, int out_size, void* d_ws, size_t ws_size,
                              hipStream_t stream) {
  (void)in_sizes; (void)n_in; (void)out_size; (void)ws_size;
  const float* x    = (const float*)d_in[0];
  const float* zhp  = (const float*)d_in[1];
  const float* lnw  = (const float*)d_in[2];
  const float* lnb  = (const float*)d_in[3];
  const float* upw  = (const float*)d_in[4];
  const float* upb  = (const float*)d_in[5];
  const float* dnw  = (const float*)d_in[6];
  const float* dnb  = (const float*)d_in[7];
  const float* encw = (const float*)d_in[8];
  const float* encb = (const float*)d_in[9];
  const float* prw  = (const float*)d_in[10];
  const float* prb  = (const float*)d_in[11];
  const float* gw   = (const float*)d_in[12];
  const float* gb   = (const float*)d_in[13];
  const float* kpw  = (const float*)d_in[14];
  const float* kpb  = (const float*)d_in[15];
  const float* vpw  = (const float*)d_in[16];
  const float* vpb  = (const float*)d_in[17];
  const float* kcw  = (const float*)d_in[18];
  const float* kcb  = (const float*)d_in[19];
  const float* vcw  = (const float*)d_in[20];
  const float* vcb  = (const float*)d_in[21];
  const float* nw   = (const float*)d_in[22];
  const float* nb   = (const float*)d_in[23];
  float* out = (float*)d_out;

  // Output tuple offsets (elements)
  const size_t O_Z = 16777216, O_ZHAT = 20971520, O_SURP = 25165824;
  const size_t O_KC = 25231360, O_VC = 33619968, O_GATE = 42008576;
  const size_t O_QN = 42074112, O_VN = 50462720, O_WN = 58851328;

  // Workspace carve-out (256B aligned)
  char* ws = (char*)d_ws;
  size_t off = 0;
  auto alloc = [&](size_t bytes) -> char* {
    char* p = ws + off;
    off = (off + bytes + 255) & ~(size_t)255;
    return p;
  };
  unsigned short* wt_enc  = (unsigned short*)alloc((size_t)CCOL * 64 * 256 * 2);
  unsigned short* wt_gain = (unsigned short*)alloc((size_t)CCOL * 256 * 64 * 2);
  unsigned short* wt_up   = (unsigned short*)alloc((size_t)CCOL * 1024 * 256 * 2);
  unsigned short* wt_dn   = (unsigned short*)alloc((size_t)CCOL * 256 * 1024 * 2);
  unsigned short* wt_pr   = (unsigned short*)alloc((size_t)CCOL * 64 * 64 * 2);
  unsigned short* wt_kp   = (unsigned short*)alloc((size_t)CCOL * 128 * 256 * 2);
  unsigned short* wt_vp   = (unsigned short*)alloc((size_t)CCOL * 128 * 256 * 2);
  unsigned short* wt_kc   = (unsigned short*)alloc((size_t)CCOL * 128 * 256 * 2);
  unsigned short* wt_vc   = (unsigned short*)alloc((size_t)CCOL * 128 * 256 * 2);
  unsigned short* xb      = (unsigned short*)alloc((size_t)CCOL * TOK * 256 * 2);
  unsigned short* actb    = (unsigned short*)alloc((size_t)CCOL * TOK * 256 * 2); // h, then x_out bf16
  unsigned short* lnxb    = (unsigned short*)alloc((size_t)CCOL * TOK * 256 * 2); // LN(x) bf16
  unsigned short* deltab  = (unsigned short*)alloc((size_t)CCOL * TOK * 64 * 2);
  unsigned short* zb      = xb;  // xb is dead after the enc GEMM; reuse for z bf16

  const dim3 blk(256, 1, 1);

  // 1) weight conversions (fp32 -> bf16 transposed)
  wconv_k<<<1024, blk, 0, stream>>>(encw, wt_enc, 256, 64);
  wconv_k<<<1024, blk, 0, stream>>>(gw,   wt_gain, 64, 256);
  wconv_k<<<2048, blk, 0, stream>>>(upw,  wt_up, 256, 1024);
  wconv_k<<<2048, blk, 0, stream>>>(dnw,  wt_dn, 1024, 256);
  wconv_k<<<256,  blk, 0, stream>>>(prw,  wt_pr, 64, 64);
  wconv_k<<<1024, blk, 0, stream>>>(kpw,  wt_kp, 256, 128);
  wconv_k<<<1024, blk, 0, stream>>>(vpw,  wt_vp, 256, 128);
  wconv_k<<<1024, blk, 0, stream>>>(kcw,  wt_kc, 256, 128);
  wconv_k<<<1024, blk, 0, stream>>>(vcw,  wt_vc, 256, 128);

  // 2) fused x conversion + LayerNorm (single read of x fp32)
  xprep_k<<<8192, blk, 0, stream>>>(x, lnw, lnb, xb, lnxb);

  // 3) z = x @ enc_w + enc_b           (K=256 -> N=64)
  gemm_k<0><<<dim3(128, 1, CCOL), blk, 0, stream>>>(
      (const unsigned int*)xb, (const unsigned int*)wt_enc, encb, 256, 64,
      out + O_Z, nullptr, nullptr);

  // 4) delta / surprise / gate
  delta_k<<<8192, blk, 0, stream>>>(out + O_Z, zhp, deltab, out + O_SURP, out + O_GATE);

  // 5) h = LN(x) * (1 + 0.1*tanh(delta @ gain_w + gain_b))   (K=64 -> N=256)
  gemm_k<1><<<dim3(128, 4, CCOL), blk, 0, stream>>>(
      (const unsigned int*)deltab, (const unsigned int*)wt_gain, gb, 64, 256,
      nullptr, actb, lnxb);

  // 6) fused FFN: x_out = x + down(gelu(up(h)))   (TDM h-tile -> LDS)
  ffn_k<<<dim3(TOK / 16, CCOL), blk, 0, stream>>>(
      (const unsigned int*)actb, (const unsigned int*)wt_up,
      (const unsigned int*)wt_dn, upb, dnb, x, out /*x_out*/, actb /*x_out bf16*/);

  // 7) z_hat = z @ pred_w + pred_b     (K=64 -> N=64)
  zconv_k<<<1024, blk, 0, stream>>>(out + O_Z, zb);
  gemm_k<0><<<dim3(128, 1, CCOL), blk, 0, stream>>>(
      (const unsigned int*)zb, (const unsigned int*)wt_pr, prb, 64, 64,
      out + O_ZHAT, nullptr, nullptr);

  // 8) fused heads: k_cand / v_cand / q_nov / v_nov  (shared A, K=256 -> N=128)
  heads_k<<<dim3(128, 2, CCOL), blk, 0, stream>>>(
      (const unsigned int*)actb,
      (const unsigned int*)wt_kp, (const unsigned int*)wt_vp,
      (const unsigned int*)wt_kc, (const unsigned int*)wt_vc,
      kpb, vpb, kcb, vcb,
      out + O_KC, out + O_VC, out + O_QN, out + O_VN);

  // 9) unit-normalize k_cand and q_nov in place
  norm_k<<<8192, blk, 0, stream>>>(out + O_KC);
  norm_k<<<8192, blk, 0, stream>>>(out + O_QN);

  // 10) novelty weight (N=1 head + sigmoid, from bf16 x_out)
  nov_k<<<8192, blk, 0, stream>>>(actb, nw, nb, out + O_WN);
}
